// SSM_73057393705055
// MI455X (gfx1250) — matrix operand order repaired and checked by reading, not compiled, not run on hardware
//
#include <hip/hip_runtime.h>
#include <hip/hip_bf16.h>

// ---------------------------------------------------------------------------
// Mamba-2 SSD forward for MI455X (gfx1250, wave32).
// Big GEMMs + chunk einsums run as bf16 WMMA (v_wmma_f32_16x16x32_bf16) with
// fp32 accumulation; scan/cumsum/exp/softplus/norm stay fp32 VALU.
// GEMM staging uses CDNA5 async global->LDS copies (ASYNCcnt) with LDS
// double-buffering so HBM latency overlaps the WMMA pipe.
// ---------------------------------------------------------------------------

#define B_ 2
#define S_ 2048
#define E_ 2048
#define H_ 64
#define P_ 64
#define N_ 128
#define K_ 4
#define CH_ 256
#define NCH 8            // S_/CH_
#define INTER 4096       // H_*P_
#define CONVD 4352       // INTER + 2*N_
#define PROJ 8512        // INTER + CONVD + H_

typedef __attribute__((ext_vector_type(16))) __bf16 v16bf;
typedef __attribute__((ext_vector_type(8)))  __bf16 v8bf;
typedef __attribute__((ext_vector_type(8)))  float  v8f;

#if defined(__has_builtin)
#  if __has_builtin(__builtin_amdgcn_global_load_async_to_lds_b128)
#    define HAVE_ASYNC_LDS 1
#  endif
#endif
#ifndef HAVE_ASYNC_LDS
#  define HAVE_ASYNC_LDS 0
#endif

// Builtin expects pointers to int4 vectors in AS(1)/AS(3) (per hipcc sema).
typedef int v4i_g __attribute__((vector_size(16)));
typedef __attribute__((address_space(1))) v4i_g* gp_b128;
typedef __attribute__((address_space(3))) v4i_g* lp_b128;

// 16B global -> LDS copy. Async (ASYNCcnt) when the toolchain exposes it.
__device__ __forceinline__ void cp_b128(const void* gsrc, void* ldst) {
#if HAVE_ASYNC_LDS
    __builtin_amdgcn_global_load_async_to_lds_b128(
        (gp_b128)(void*)gsrc, (lp_b128)ldst, 0, 0);
#else
    *(uint4*)ldst = *(const uint4*)gsrc;
#endif
}

template <int Ncnt>
__device__ __forceinline__ void wait_async() {
#if HAVE_ASYNC_LDS
#  if __has_builtin(__builtin_amdgcn_s_wait_asynccnt)
    __builtin_amdgcn_s_wait_asynccnt((unsigned short)Ncnt);
#  else
    asm volatile("s_wait_asynccnt %0" ::"i"(Ncnt) : "memory");
#  endif
#endif
}

// ---------------- WMMA fragment helpers (layouts per ISA 7.12.2) -----------
// A fragment 16x32 bf16: lane = {khalf[1], m[4]}; elems 0..7 = K kb..kb+7,
// elems 8..15 = K kb+16..kb+23, kb = (lane>=16)?8:0. Row-major src, stride ld.
__device__ inline v16bf load_a_frag(const __bf16* __restrict__ src, int ld) {
    int lane = threadIdx.x & 31;
    int m  = lane & 15;
    int kb = (lane >> 4) << 3;
    const v8bf* p0 = (const v8bf*)(src + (size_t)m * ld + kb);
    const v8bf* p1 = (const v8bf*)(src + (size_t)m * ld + kb + 16);
    v8bf lo = *p0, hi = *p1;
    v16bf r;
#pragma unroll
    for (int i = 0; i < 8; ++i) { r[i] = lo[i]; r[i + 8] = hi[i]; }
    return r;
}

// B fragment 32x16 bf16: lane col n = lane&15; elems 0..7 = K kb..kb+7,
// elems 8..15 = K kb+16..kb+23, kb = (lane>=16)?8:0. src is [n][k] row-major (k contiguous), stride ld.
__device__ inline v16bf load_b_frag(const __bf16* __restrict__ src, int ld) {
    int lane = threadIdx.x & 31;
    int n  = lane & 15;
    int kb = (lane >> 4) << 3;
    const v8bf* p0 = (const v8bf*)(src + (size_t)n * ld + kb);
    const v8bf* p1 = (const v8bf*)(src + (size_t)n * ld + kb + 16);
    v8bf lo = *p0, hi = *p1;
    v16bf r;
#pragma unroll
    for (int i = 0; i < 8; ++i) { r[i] = lo[i]; r[i + 8] = hi[i]; }
    return r;
}

__device__ inline v8f wmma_bf16(v16bf a, v16bf b, v8f c) {
    return __builtin_amdgcn_wmma_f32_16x16x32_bf16(false, a, false, b,
                                                   (short)0, c, false, false);
}

__device__ inline float silu_f(float x) { return x / (1.f + __expf(-x)); }

// ---------------- fp32 -> bf16 convert -------------------------------------
__global__ void f2bf_kernel(const float* __restrict__ src,
                            __bf16* __restrict__ dst, size_t n) {
    size_t i = (size_t)blockIdx.x * blockDim.x + threadIdx.x;
    if (i < n) dst[i] = (__bf16)src[i];
}

// ---------------- bf16 WMMA GEMM: C[M,N] = A[M,K] * W[N,K]^T ----------------
// Block tile 256x64, 8 waves in 4x2, each wave 64x32 (4x2 wmma tiles).
// Double-buffered LDS; async global->LDS staging overlaps the WMMA pipe.
// Requires M%256==0, N%64==0, K%32==0 (holds for both GEMMs here).
#define BM_G 256
#define BN_G 64
#define LDA_S 48
__global__ __launch_bounds__(256) void gemm_bf16_nt(
    const __bf16* __restrict__ A, const __bf16* __restrict__ W,
    float* __restrict__ C, int M, int N, int K) {
    __shared__ __bf16 As[2][BM_G * LDA_S];
    __shared__ __bf16 Ws[2][BN_G * LDA_S];
    int m0 = blockIdx.y * BM_G;
    int n0 = blockIdx.x * BN_G;
    int tid = threadIdx.x;
    int wave = tid >> 5, lane = tid & 31;
    int mw = (wave >> 1) * 64;
    int nw = (wave & 1) * 32;
    v8f acc[4][2] = {};
    int rsu = K >> 3;  // global row stride in uint4
    const uint4* gA = (const uint4*)(A + (size_t)m0 * K);
    const uint4* gW = (const uint4*)(W + (size_t)n0 * K);

    // stage one 32-wide K tile into buffer `buf` (5 x 16B per thread)
    auto stage = [&](int buf, int kt) {
        int kq4 = kt * 4;
#pragma unroll
        for (int i = 0; i < 4; ++i) {          // A: 256 rows x 4 uint4
            int lin = tid + i * 256;
            int row = lin >> 2, kq = lin & 3;
            cp_b128(&gA[(size_t)row * rsu + kq4 + kq],
                    &As[buf][row * LDA_S + kq * 8]);
        }
        {                                       // W: 64 rows x 4 uint4
            int row = tid >> 2, kq = tid & 3;
            cp_b128(&gW[(size_t)row * rsu + kq4 + kq],
                    &Ws[buf][row * LDA_S + kq * 8]);
        }
    };

    int nk = K >> 5;
    stage(0, 0);
    for (int kt = 0; kt < nk; ++kt) {
        int cur = kt & 1;
        if (kt + 1 < nk) {
            stage(cur ^ 1, kt + 1);   // prefetch next tile into other buffer
            wait_async<5>();          // wait for current tile only
        } else {
            wait_async<0>();
        }
        __syncthreads();
        v16bf b0 = load_b_frag(&Ws[cur][(nw + 0) * LDA_S], LDA_S);
        v16bf b1 = load_b_frag(&Ws[cur][(nw + 16) * LDA_S], LDA_S);
#pragma unroll
        for (int mt = 0; mt < 4; ++mt) {
            v16bf a = load_a_frag(&As[cur][(mw + mt * 16) * LDA_S], LDA_S);
            acc[mt][0] = wmma_bf16(a, b0, acc[mt][0]);
            acc[mt][1] = wmma_bf16(a, b1, acc[mt][1]);
        }
        __syncthreads();
    }
    int col = lane & 15, rb = (lane >> 4) * 8;
#pragma unroll
    for (int mt = 0; mt < 4; ++mt)
#pragma unroll
        for (int nt = 0; nt < 2; ++nt)
#pragma unroll
            for (int r = 0; r < 8; ++r)
                C[(size_t)(m0 + mw + mt * 16 + rb + r) * N +
                  (n0 + nw + nt * 16 + col)] = acc[mt][nt][r];
}

// ---------------- depthwise causal conv (K=4) + SiLU -> bf16 ----------------
__global__ __launch_bounds__(256) void conv_silu_kernel(
    const float* __restrict__ proj, const float* __restrict__ cw,
    const float* __restrict__ cb, __bf16* __restrict__ hBC) {
    int s = blockIdx.x, b = blockIdx.y, tid = threadIdx.x;
    size_t row = (size_t)b * S_ + s;
#pragma unroll
    for (int j = 0; j < CONVD / 256; ++j) {
        int ch = tid + j * 256;
        float acc = cb[ch];
#pragma unroll
        for (int k = 0; k < K_; ++k) {
            int ss = s - (K_ - 1) + k;
            if (ss >= 0)
                acc += proj[((size_t)b * S_ + ss) * PROJ + INTER + ch] *
                       cw[ch * K_ + k];
        }
        hBC[row * CONVD + ch] = (__bf16)silu_f(acc);
    }
}

// ---------------- dt: softplus(dt+bias), Adt = -exp(A_log)*dt ---------------
__global__ void dt_kernel(const float* __restrict__ proj,
                          const float* __restrict__ dtb,
                          const float* __restrict__ Alog,
                          float* __restrict__ dtS, float* __restrict__ AdtG) {
    int idx = blockIdx.x * blockDim.x + threadIdx.x;
    int h = idx & (H_ - 1);
    size_t row = (size_t)(idx >> 6);
    float x = proj[row * PROJ + INTER + CONVD + h] + dtb[h];
    float sp = (x > 20.f) ? x : log1pf(__expf(x));
    dtS[idx]  = sp;
    AdtG[idx] = -__expf(Alog[h]) * sp;
}

// ---------------- SSD chunk kernel: per (b, chunk, head) --------------------
// Computes A_cum (scan), intra-chunk states (WMMA), causal Y_diag (WMMA) + D.
#define XLD 264  // padded LDS stride for 256-long K rows
__global__ __launch_bounds__(256) void ssd_chunk_kernel(
    const __bf16* __restrict__ hBC,  // [B,S,CONVD] bf16 (post conv+silu)
    const float* __restrict__ dtS,   // [B,S,H]
    const float* __restrict__ AdtG,  // [B,S,H]
    const float* __restrict__ Dp,    // [H]
    float* __restrict__ AcumG,       // [B,H,NCH,CH]
    float* __restrict__ statesG,     // [B,NCH,H,N,P] fp32
    float* __restrict__ yG)          // [B,S,H*P]
{
    extern __shared__ __align__(16) char smem_raw[];
    float*  Acum = (float*)smem_raw;                 // 256
    float*  dec  = Acum + 256;                       // 256
    float*  dtl  = dec + 256;                        // 256
    __bf16* xcT  = (__bf16*)(dtl + 256);             // 64  x XLD  (x*dt)^T
    __bf16* BwT  = xcT + 64 * XLD;                   // 128 x XLD  (B*decay)^T
    __bf16* Mb   = BwT + 128 * XLD;                  // 8 waves x 16x32

    int h = blockIdx.x, c = blockIdx.y, b = blockIdx.z;
    int tid = threadIdx.x, lane = tid & 31, wave = tid >> 5;
    size_t rowBase = (size_t)b * S_ + c * CH_;

    Acum[tid] = AdtG[(rowBase + tid) * H_ + h];
    dtl[tid]  = dtS[(rowBase + tid) * H_ + h];
    __syncthreads();
    // inclusive Hillis-Steele scan over 256 elements
    for (int off = 1; off < 256; off <<= 1) {
        float v = (tid >= off) ? Acum[tid - off] : 0.f;
        __syncthreads();
        Acum[tid] += v;
        __syncthreads();
    }
    float Alast = Acum[255];
    dec[tid] = __expf(Alast - Acum[tid]);
    AcumG[(((size_t)b * H_ + h) * NCH + c) * CH_ + tid] = Acum[tid];
    __syncthreads();

    const __bf16* hsBase = hBC + rowBase * CONVD;
    // stage xcT[p][l] = hs[l][h*64+p] * dt[l]
    for (int idx = tid; idx < 64 * 256; idx += 256) {
        int p = idx & 63, l = idx >> 6;
        float v = (float)hsBase[(size_t)l * CONVD + h * P_ + p] * dtl[l];
        xcT[p * XLD + l] = (__bf16)v;
    }
    // stage BwT[n][l] = B[l][n] * exp(Alast - Acum[l])
    for (int idx = tid; idx < 128 * 256; idx += 256) {
        int n = idx & 127, l = idx >> 7;
        float v = (float)hsBase[(size_t)l * CONVD + INTER + n] * dec[l];
        BwT[n * XLD + l] = (__bf16)v;
    }
    __syncthreads();

    int colp = lane & 15, rb = (lane >> 4) * 8;

    // ---- states[n][p] = sum_l Bw[l][n] * xc[l][p]  (128x64, K=256) ----
    {
        int mt = wave;  // 8 waves -> 8 n-tiles of 16
        v8f acc[4] = {};
        for (int kk = 0; kk < 8; ++kk) {
            v16bf a = load_a_frag(&BwT[(mt * 16) * XLD + kk * 32], XLD);
#pragma unroll
            for (int nt = 0; nt < 4; ++nt) {
                v16bf bx = load_b_frag(&xcT[(nt * 16) * XLD + kk * 32], XLD);
                acc[nt] = wmma_bf16(a, bx, acc[nt]);
            }
        }
        float* sg = statesG + (((size_t)b * NCH + c) * H_ + h) * (N_ * P_);
#pragma unroll
        for (int nt = 0; nt < 4; ++nt)
#pragma unroll
            for (int r = 0; r < 8; ++r)
                sg[(mt * 16 + rb + r) * P_ + nt * 16 + colp] = acc[nt][r];
    }

    // ---- Y_diag: causal (C B^T * L) @ xc, + D residual ----
    const __bf16* Cbase = hsBase + INTER + N_;
    const __bf16* Bbase = hsBase + INTER;
    __bf16* Mw = Mb + wave * (16 * 32);
    float Dh = Dp[h];
    for (int pass = 0; pass < 2; ++pass) {
        int lt = (pass == 0) ? wave : 15 - wave;   // load-balanced l-tiles
        v8f accy[4] = {};
        for (int sp = 0; sp <= (lt >> 1); ++sp) {
#pragma unroll
            for (int half = 0; half < 2; ++half) {
                int st = sp * 2 + half;
                if (st <= lt) {
                    v8f t = {};
                    for (int kk = 0; kk < 4; ++kk) {  // K = N_ = 128
                        v16bf a = load_a_frag(
                            Cbase + (size_t)(lt * 16) * CONVD + kk * 32, CONVD);
                        v16bf bb = load_b_frag(
                            Bbase + (size_t)(st * 16) * CONVD + kk * 32, CONVD);
                        t = wmma_bf16(a, bb, t);
                    }
#pragma unroll
                    for (int r = 0; r < 8; ++r) {
                        int l = lt * 16 + rb + r;
                        int s = st * 16 + colp;
                        float v = (s <= l) ? t[r] * __expf(Acum[l] - Acum[s]) : 0.f;
                        Mw[(rb + r) * 32 + half * 16 + colp] = (__bf16)v;
                    }
                } else {
#pragma unroll
                    for (int r = 0; r < 8; ++r)
                        Mw[(rb + r) * 32 + half * 16 + colp] = (__bf16)0.f;
                }
            }
            v16bf am = load_a_frag(Mw, 32);
#pragma unroll
            for (int nt = 0; nt < 4; ++nt) {
                v16bf bx = load_b_frag(&xcT[(nt * 16) * XLD + sp * 32], XLD);
                accy[nt] = wmma_bf16(am, bx, accy[nt]);
            }
        }
#pragma unroll
        for (int nt = 0; nt < 4; ++nt)
#pragma unroll
            for (int r = 0; r < 8; ++r) {
                int l = lt * 16 + rb + r;
                int p = nt * 16 + colp;
                float hs = (float)hsBase[(size_t)l * CONVD + h * P_ + p];
                yG[(rowBase + l) * INTER + h * P_ + p] = accy[nt][r] + Dh * hs;
            }
    }
}

// ---------------- inter-chunk recurrence (tiny, fp32) -----------------------
__global__ __launch_bounds__(256) void chunk_rec_kernel(
    const float* __restrict__ AcumG, const float* __restrict__ statesG,
    __bf16* __restrict__ prevT) {   // prevT: [B,NCH,H,P,N] bf16
    int h = blockIdx.x, b = blockIdx.y, tid = threadIdx.x;
    __shared__ float tc[NCH];
    if (tid < NCH)
        tc[tid] = AcumG[(((size_t)b * H_ + h) * NCH + tid) * CH_ + (CH_ - 1)];
    __syncthreads();
    float run[32];
#pragma unroll
    for (int k = 0; k < 32; ++k) run[k] = 0.f;
    for (int c = 0; c < NCH; ++c) {
        const float* sg = statesG + (((size_t)b * NCH + c) * H_ + h) * (N_ * P_);
        __bf16* pg = prevT + (((size_t)b * NCH + c) * H_ + h) * (N_ * P_);
        float e = __expf(tc[c]);
#pragma unroll
        for (int k = 0; k < 32; ++k) {
            int idx = tid + k * 256;        // idx = n*64 + p
            int n = idx >> 6, p = idx & 63;
            pg[p * N_ + n] = (__bf16)run[k];        // transpose to [p][n]
            run[k] = run[k] * e + sg[idx];
        }
    }
}

// ---------------- Y_off += state_decay * C @ prev_states^T ------------------
__global__ __launch_bounds__(256) void y_off_kernel(
    const __bf16* __restrict__ hBC, const __bf16* __restrict__ prevT,
    const float* __restrict__ AcumG, float* __restrict__ yG) {
    __shared__ float sd[256];
    int h = blockIdx.x, c = blockIdx.y, b = blockIdx.z;
    int tid = threadIdx.x, lane = tid & 31, wave = tid >> 5;
    sd[tid] = __expf(AcumG[(((size_t)b * H_ + h) * NCH + c) * CH_ + tid]);
    __syncthreads();
    size_t rowBase = (size_t)b * S_ + c * CH_;
    const __bf16* Cbase = hBC + rowBase * CONVD + INTER + N_;
    const __bf16* pv = prevT + (((size_t)b * NCH + c) * H_ + h) * (N_ * P_);
    int colp = lane & 15, rb = (lane >> 4) * 8;
    for (int pass = 0; pass < 2; ++pass) {
        int lt = wave + pass * 8;
        v8f acc[4] = {};
        for (int kk = 0; kk < 4; ++kk) {    // K = N_ = 128
            v16bf a = load_a_frag(Cbase + (size_t)(lt * 16) * CONVD + kk * 32,
                                  CONVD);
#pragma unroll
            for (int nt = 0; nt < 4; ++nt) {
                v16bf bp = load_b_frag(pv + (nt * 16) * N_ + kk * 32, N_);
                acc[nt] = wmma_bf16(a, bp, acc[nt]);
            }
        }
#pragma unroll
        for (int nt = 0; nt < 4; ++nt)
#pragma unroll
            for (int r = 0; r < 8; ++r) {
                int l = lt * 16 + rb + r;
                int p = nt * 16 + colp;
                float* yp = &yG[(rowBase + l) * INTER + h * P_ + p];
                *yp += sd[l] * acc[nt][r];
            }
    }
}

// ---------------- gated RMSNorm -> bf16 -------------------------------------
__global__ __launch_bounds__(256) void gate_norm_kernel(
    const float* __restrict__ yG, const float* __restrict__ proj,
    const float* __restrict__ nw, __bf16* __restrict__ yn) {
    __shared__ float red[256];
    size_t row = blockIdx.x;
    int tid = threadIdx.x;
    const float* yr = yG + row * INTER;
    const float* gr = proj + row * PROJ;
    float v[16];
    float ss = 0.f;
#pragma unroll
    for (int k = 0; k < 16; ++k) {
        int i = tid + k * 256;
        float t = yr[i] * silu_f(gr[i]);
        v[k] = t;
        ss += t * t;
    }
    red[tid] = ss;
    __syncthreads();
    for (int off = 128; off > 0; off >>= 1) {
        if (tid < off) red[tid] += red[tid + off];
        __syncthreads();
    }
    float r = rsqrtf(red[0] / (float)INTER + 1e-5f);
#pragma unroll
    for (int k = 0; k < 16; ++k) {
        int i = tid + k * 256;
        yn[row * INTER + i] = (__bf16)(nw[i] * v[k] * r);
    }
}

// ---------------------------------------------------------------------------
extern "C" void kernel_launch(void* const* d_in, const int* in_sizes, int n_in,
                              void* d_out, int out_size, void* d_ws,
                              size_t ws_size, hipStream_t stream) {
    (void)in_sizes; (void)n_in; (void)out_size; (void)ws_size;
    const float* x    = (const float*)d_in[0];
    // d_in[1] = mask (all ones, unused by reference math)
    const float* Win  = (const float*)d_in[2];
    const float* cw   = (const float*)d_in[3];
    const float* cb   = (const float*)d_in[4];
    const float* Alog = (const float*)d_in[5];
    const float* dtb  = (const float*)d_in[6];
    const float* Dp   = (const float*)d_in[7];
    const float* nw   = (const float*)d_in[8];
    const float* Wout = (const float*)d_in[9];
    float* out = (float*)d_out;

    char* ws = (char*)d_ws;
    size_t off = 0;
    auto alloc = [&](size_t bytes) -> char* {
        char* p = ws + off;
        off += (bytes + 255) & ~(size_t)255;
        return p;
    };
    const size_t nX   = (size_t)B_ * S_ * E_;
    const size_t nWin = (size_t)PROJ * E_;
    const size_t nWo  = (size_t)E_ * INTER;
    __bf16* xb     = (__bf16*)alloc(nX * 2);
    __bf16* Winb   = (__bf16*)alloc(nWin * 2);
    __bf16* Woutb  = (__bf16*)alloc(nWo * 2);
    float*  proj   = (float*)alloc((size_t)B_ * S_ * PROJ * 4);
    __bf16* hBCb   = (__bf16*)alloc((size_t)B_ * S_ * CONVD * 2);
    float*  dtS    = (float*)alloc((size_t)B_ * S_ * H_ * 4);
    float*  AdtG   = (float*)alloc((size_t)B_ * S_ * H_ * 4);
    float*  AcumG  = (float*)alloc((size_t)B_ * H_ * NCH * CH_ * 4);
    float*  states = (float*)alloc((size_t)B_ * NCH * H_ * N_ * P_ * 4);
    __bf16* prevT  = (__bf16*)alloc((size_t)B_ * NCH * H_ * N_ * P_ * 2);
    float*  yG     = (float*)alloc((size_t)B_ * S_ * INTER * 4);
    __bf16* ynb    = (__bf16*)alloc((size_t)B_ * S_ * INTER * 2);

    // 1) convert to bf16
    f2bf_kernel<<<(nX + 255) / 256, 256, 0, stream>>>(x, xb, nX);
    f2bf_kernel<<<(nWin + 255) / 256, 256, 0, stream>>>(Win, Winb, nWin);
    f2bf_kernel<<<(nWo + 255) / 256, 256, 0, stream>>>(Wout, Woutb, nWo);

    // 2) in_proj GEMM: [4096,8512] = x[4096,2048] @ Win[8512,2048]^T
    gemm_bf16_nt<<<dim3(PROJ / BN_G, (B_ * S_) / BM_G), 256, 0, stream>>>(
        xb, Winb, proj, B_ * S_, PROJ, E_);

    // 3) conv + SiLU, dt transforms
    conv_silu_kernel<<<dim3(S_, B_), 256, 0, stream>>>(proj, cw, cb, hBCb);
    dt_kernel<<<(B_ * S_ * H_) / 256, 256, 0, stream>>>(proj, dtb, Alog, dtS,
                                                        AdtG);

    // 4) SSD chunk: states + Y_diag (dynamic LDS ~112.6 KB)
    size_t ssd_lds = 3 * 256 * 4 + (64 * XLD + 128 * XLD + 8 * 16 * 32) * 2;
    ssd_chunk_kernel<<<dim3(H_, NCH, B_), 256, ssd_lds, stream>>>(
        hBCb, dtS, AdtG, Dp, AcumG, states, yG);

    // 5) inter-chunk recurrence -> prev states (bf16, transposed [p][n])
    chunk_rec_kernel<<<dim3(H_, B_), 256, 0, stream>>>(AcumG, states, prevT);

    // 6) Y_off accumulate
    y_off_kernel<<<dim3(H_, NCH, B_), 256, 0, stream>>>(hBCb, prevT, AcumG, yG);

    // 7) gated RMSNorm -> bf16
    gate_norm_kernel<<<B_ * S_, 256, 0, stream>>>(yG, proj, nw, ynb);

    // 8) out_proj GEMM: out[4096,2048] = yn[4096,4096] @ Wout[2048,4096]^T
    gemm_bf16_nt<<<dim3(E_ / BN_G, (B_ * S_) / BM_G), 256, 0, stream>>>(
        ynb, Woutb, out, B_ * S_, E_, INTER);
}
